// CausalSelfAttention_26946624815585
// MI455X (gfx1250) — matrix operand order repaired and checked by reading, not compiled, not run on hardware
//
#include <hip/hip_runtime.h>
#include <hip/hip_bf16.h>

typedef __attribute__((ext_vector_type(16))) __bf16 bf16x16;
typedef __attribute__((ext_vector_type(8)))  __bf16 bf16x8;
typedef __attribute__((ext_vector_type(8)))  float  f32x8;

// Native conversions: clang emits gfx1250 bf16 cvt ops (round-to-nearest-even)
__device__ __forceinline__ __bf16 f2bf(float f) { return (__bf16)f; }
__device__ __forceinline__ float bf2f(__bf16 b) { return (float)b; }
__device__ __forceinline__ f32x8 f32x8_zero() {
  f32x8 v = {0.f,0.f,0.f,0.f,0.f,0.f,0.f,0.f};
  return v;
}

// ---------------------------------------------------------------------------
// Generic bf16 WMMA GEMM: C[M,N] = A[M,K] * W[N,K]^T
// Block tile 128x128, 256 threads (8 waves in 2x4), K-step 32.
// LDS tiles are stored in WMMA *fragment order* so each lane's fragment load
// is a single contiguous 32-byte ds read.
//   A frag (16x32, 16-bit): lane = (m&15)+16*g ; element e -> k:
//     k = 16*((e>>3)&1) + 8*g + (e&7)   [inverse: g=(k>>3)&1, e=8*(k>>4)+(k&7)]
//   B frag (32x16, 16-bit): lane = (n&15)+16*g ; element e -> k = 16*((e>>3)&1) + 8*g + (e&7) (as A)
//   C/D frag (16x16 f32):   lane: n = lane&15, row = r + 8*(lane>>4)
// EPI==0: store f32 C.  EPI==1: scatter bf16 into Q[b,h,s,d], K[b,h,s,d],
//         Vt[b,h,d,s] (d_model split 3*16*64 along N).
// ---------------------------------------------------------------------------
template<int EPI, bool ABF>
__global__ __launch_bounds__(256, 1)
void gemm_bf16_wmma(const float* __restrict__ A32, const __bf16* __restrict__ A16,
                    const float* __restrict__ W, int K,
                    float* __restrict__ C, int ldc,
                    __bf16* __restrict__ Qp, __bf16* __restrict__ Kp,
                    __bf16* __restrict__ Vt)
{
  __shared__ __bf16 Abuf[128 * 32];   // 8 KB, fragment-ordered
  __shared__ __bf16 Bbuf[128 * 32];   // 8 KB, fragment-ordered

  const int tid  = threadIdx.x;
  const int lane = tid & 31;
  const int wid  = tid >> 5;
  const int wm   = wid >> 2;          // 0..1  (M wave row)
  const int wn   = wid & 3;           // 0..3  (N wave col)
  const int g    = lane >> 4;
  const int ln   = lane & 15;
  const int blockM = blockIdx.x * 128;
  const int blockN = blockIdx.y * 128;

  f32x8 acc[4][2];
  #pragma unroll
  for (int i = 0; i < 4; ++i)
    #pragma unroll
    for (int j = 0; j < 2; ++j)
      acc[i][j] = f32x8_zero();

  for (int kb = 0; kb < K; kb += 32) {
    __syncthreads();
    // ---- stage A and B tiles into LDS (2 chunks of 8 elements each) ----
    #pragma unroll
    for (int cc = 0; cc < 2; ++cc) {
      int c    = tid + cc * 256;      // 0..511
      int rLoc = c >> 2;              // 0..127 row within tile
      int gg   = c & 1;               // lane group
      int hsel = (c >> 1) & 1;        // which 8-element half
      // A: k = kb + 16*hsel + 8*gg + (0..7) ; e0 = 8*hsel
      {
        int kk = kb + 16 * hsel + 8 * gg;
        bf16x8 v;
        if constexpr (ABF) {
          const __bf16* p = A16 + (size_t)(blockM + rLoc) * 1024 + kk;
          v = *(const bf16x8*)p;
          if (kb + 32 < K) __builtin_prefetch(p + 32, 0, 3);  // next K tile
        } else {
          const float* p = A32 + (size_t)(blockM + rLoc) * 1024 + kk;
          float4 t0 = *(const float4*)p;
          float4 t1 = *(const float4*)(p + 4);
          v[0]=f2bf(t0.x); v[1]=f2bf(t0.y); v[2]=f2bf(t0.z); v[3]=f2bf(t0.w);
          v[4]=f2bf(t1.x); v[5]=f2bf(t1.y); v[6]=f2bf(t1.z); v[7]=f2bf(t1.w);
          if (kb + 32 < K) __builtin_prefetch(p + 32, 0, 3);  // next K tile
        }
        int frag = rLoc >> 4, laneL = (rLoc & 15) + 16 * gg;
        *(bf16x8*)&Abuf[frag * 512 + laneL * 16 + 8 * hsel] = v;
      }
      // B: k = kb + 16*hsel + 8*gg + (0..7) ; e0 = 8*hsel
      {
        int kk = kb + 16 * hsel + 8 * gg;
        const float* p = W + (size_t)(blockN + rLoc) * 1024 + kk;
        float4 t0 = *(const float4*)p;
        float4 t1 = *(const float4*)(p + 4);
        bf16x8 v;
        v[0]=f2bf(t0.x); v[1]=f2bf(t0.y); v[2]=f2bf(t0.z); v[3]=f2bf(t0.w);
        v[4]=f2bf(t1.x); v[5]=f2bf(t1.y); v[6]=f2bf(t1.z); v[7]=f2bf(t1.w);
        if (kb + 32 < K) __builtin_prefetch(p + 32, 0, 3);    // next K tile
        int frag = rLoc >> 4, laneL = (rLoc & 15) + 16 * gg;
        *(bf16x8*)&Bbuf[frag * 512 + laneL * 16 + 8 * hsel] = v;
      }
    }
    __syncthreads();

    // ---- fragment loads + 8 WMMAs per wave per K step ----
    bf16x16 af[4], bw[2];
    #pragma unroll
    for (int i = 0; i < 4; ++i)
      af[i] = *(const bf16x16*)&Abuf[(wm * 4 + i) * 512 + lane * 16];
    #pragma unroll
    for (int j = 0; j < 2; ++j)
      bw[j] = *(const bf16x16*)&Bbuf[(wn * 2 + j) * 512 + lane * 16];
    #pragma unroll
    for (int i = 0; i < 4; ++i)
      #pragma unroll
      for (int j = 0; j < 2; ++j)
        acc[i][j] = __builtin_amdgcn_wmma_f32_16x16x32_bf16(
            false, af[i], false, bw[j], (short)0, acc[i][j], false, false);
  }

  // ---- epilogue ----
  #pragma unroll
  for (int i = 0; i < 4; ++i) {
    #pragma unroll
    for (int j = 0; j < 2; ++j) {
      int col = blockN + (wn * 2 + j) * 16 + ln;
      if constexpr (EPI == 0) {
        #pragma unroll
        for (int r = 0; r < 8; ++r) {
          int row = blockM + (wm * 4 + i) * 16 + r + 8 * g;
          C[(size_t)row * ldc + col] = acc[i][j][r];
        }
      } else {
        int which = col >> 10;          // 0=Q 1=K 2=V (uniform per block)
        int h = (col >> 6) & 15, d = col & 63;
        #pragma unroll
        for (int r = 0; r < 8; ++r) {
          int row = blockM + (wm * 4 + i) * 16 + r + 8 * g;
          int b = row >> 11, s = row & 2047;
          int bh = b * 16 + h;
          __bf16 v = f2bf(acc[i][j][r]);
          if (which == 0)      Qp[((size_t)bh * 2048 + s) * 64 + d] = v;
          else if (which == 1) Kp[((size_t)bh * 2048 + s) * 64 + d] = v;
          else                 Vt[((size_t)bh * 64 + d) * 2048 + s] = v;
        }
      }
    }
  }
}

// ---------------------------------------------------------------------------
// RoPE in-place on Q and K, layout [b,h,s,64]. One thread per (token, j<32).
// ---------------------------------------------------------------------------
__global__ __launch_bounds__(256, 1)
void rope_kernel(__bf16* __restrict__ Q, __bf16* __restrict__ K)
{
  int idx = blockIdx.x * 256 + threadIdx.x;   // < B*H*S*32
  int j  = idx & 31;
  int tk = idx >> 5;                          // bh*2048 + s
  int s  = tk & 2047;
  // inv_freq = 10000^(-j/32) = exp(-j * ln(10000)/32)
  float inv = __expf(-(float)j * 0.28782313662425572f);
  float ang = (float)s * inv;
  float sn, cs;
  __sincosf(ang, &sn, &cs);
  size_t base = (size_t)tk * 64 + j;
  float q1 = bf2f(Q[base]), q2 = bf2f(Q[base + 32]);
  Q[base]      = f2bf(q1 * cs - q2 * sn);
  Q[base + 32] = f2bf(q2 * cs + q1 * sn);
  float k1 = bf2f(K[base]), k2 = bf2f(K[base + 32]);
  K[base]      = f2bf(k1 * cs - k2 * sn);
  K[base + 32] = f2bf(k2 * cs + k1 * sn);
}

// ---------------------------------------------------------------------------
// Flash attention: grid (B*H, S/64), 128 threads (4 waves x 16 q-rows).
// Per 32-key step: 4 WMMAs for Q*K^T, f32 online softmax (shfl_xor within the
// 16-lane halves), P shuffled C-layout -> A-layout through LDS, 4 WMMAs P*V.
// Output written bf16 into AO[b*2048+s][h*64+d] for the out-proj GEMM.
// ---------------------------------------------------------------------------
__global__ __launch_bounds__(128, 1)
void attn_kernel(const __bf16* __restrict__ Q, const __bf16* __restrict__ K,
                 const __bf16* __restrict__ Vt, __bf16* __restrict__ AO)
{
  __shared__ __bf16 ldsP[4 * 512];            // 1 KB per wave

  const int bh    = blockIdx.x;               // 0..63
  const int qtile = blockIdx.y;               // 0..31
  const int tid   = threadIdx.x;
  const int lane  = tid & 31;
  const int wid   = tid >> 5;
  const int g     = lane >> 4;
  const int ln    = lane & 15;
  const int qBase = qtile * 64 + wid * 16;

  // Q A-fragments (16 rows x 32 of D), held for the whole loop.
  const __bf16* Qrow = Q + ((size_t)bh * 2048 + qBase + ln) * 64;
  bf16x16 qa[2];
  #pragma unroll
  for (int kc = 0; kc < 2; ++kc) {
    bf16x8 lo = *(const bf16x8*)(Qrow + kc * 32 + g * 16);
    bf16x8 hi = *(const bf16x8*)(Qrow + kc * 32 + g * 16 + 8);
    qa[kc] = __builtin_shufflevector(lo, hi, 0,1,2,3,4,5,6,7,8,9,10,11,12,13,14,15);
  }

  f32x8 o[4];
  float mi[8], li[8];
  #pragma unroll
  for (int i = 0; i < 4; ++i) o[i] = f32x8_zero();
  #pragma unroll
  for (int r = 0; r < 8; ++r) { mi[r] = -3.0e38f; li[r] = 0.f; }

  const int kbMax = qtile * 2 + 2;            // block-uniform causal bound
  for (int kb = 0; kb < kbMax; ++kb) {
    // ---- scores: S = Q * K^T (16 x 32 tile) ----
    f32x8 sacc[2] = { f32x8_zero(), f32x8_zero() };
    #pragma unroll
    for (int nf = 0; nf < 2; ++nf) {
      const __bf16* Kr =
          K + ((size_t)bh * 2048 + kb * 32 + nf * 16 + ln) * 64 + g * 16;
      #pragma unroll
      for (int kc = 0; kc < 2; ++kc) {
        bf16x16 kf = *(const bf16x16*)(Kr + kc * 32);
        sacc[nf] = __builtin_amdgcn_wmma_f32_16x16x32_bf16(
            false, qa[kc], false, kf, (short)0, sacc[nf], false, false);
      }
    }

    // ---- causal mask, scale, online softmax ----
    float alpha[8];
    #pragma unroll
    for (int r = 0; r < 8; ++r) {
      const int qg = qBase + r + 8 * g;
      #pragma unroll
      for (int nf = 0; nf < 2; ++nf) {
        int kg = kb * 32 + nf * 16 + ln;
        float v = sacc[nf][r] * 0.125f;       // 1/sqrt(64)
        sacc[nf][r] = (kg <= qg) ? v : -3.0e38f;
      }
      float mx = fmaxf(sacc[0][r], sacc[1][r]);
      #pragma unroll
      for (int off = 1; off < 16; off <<= 1)
        mx = fmaxf(mx, __shfl_xor(mx, off, 32));
      float nm = fmaxf(mi[r], mx);
      float p0 = __expf(sacc[0][r] - nm);
      float p1 = __expf(sacc[1][r] - nm);
      sacc[0][r] = p0; sacc[1][r] = p1;
      float sum = p0 + p1;
      #pragma unroll
      for (int off = 1; off < 16; off <<= 1)
        sum += __shfl_xor(sum, off, 32);
      alpha[r] = __expf(mi[r] - nm);
      li[r] = li[r] * alpha[r] + sum;
      mi[r] = nm;
    }
    #pragma unroll
    for (int df = 0; df < 4; ++df)
      #pragma unroll
      for (int r = 0; r < 8; ++r) o[df][r] *= alpha[r];

    // ---- P: C-layout -> A-fragment layout through LDS ----
    #pragma unroll
    for (int nf = 0; nf < 2; ++nf)
      #pragma unroll
      for (int r = 0; r < 8; ++r) {
        int kcol = nf * 16 + ln;
        int m    = r + 8 * g;
        int gA   = (kcol >> 4) & 1;
        int eA   = kcol & 15;
        ldsP[wid * 512 + (m + 16 * gA) * 16 + eA] = f2bf(sacc[nf][r]);
      }
    __syncthreads();
    bf16x16 pA = *(const bf16x16*)&ldsP[wid * 512 + lane * 16];

    // ---- O += P * V  (V stored transposed [b,h,d,s] -> contiguous frags) ----
    #pragma unroll
    for (int df = 0; df < 4; ++df) {
      const __bf16* Vr =
          Vt + ((size_t)bh * 64 + df * 16 + ln) * 2048 + kb * 32 + g * 16;
      bf16x16 vf = *(const bf16x16*)Vr;
      o[df] = __builtin_amdgcn_wmma_f32_16x16x32_bf16(
          false, pA, false, vf, (short)0, o[df], false, false);
    }
    __syncthreads();
  }

  // ---- normalize and write bf16 to AO[token][h*64+d] ----
  const int b = bh >> 4, h = bh & 15;
  #pragma unroll
  for (int df = 0; df < 4; ++df)
    #pragma unroll
    for (int r = 0; r < 8; ++r) {
      int s = qBase + r + 8 * g;
      float v = o[df][r] / li[r];
      AO[((size_t)(b * 2048 + s)) * 1024 + h * 64 + df * 16 + ln] = f2bf(v);
    }
}

// ---------------------------------------------------------------------------
extern "C" void kernel_launch(void* const* d_in, const int* in_sizes, int n_in,
                              void* d_out, int out_size, void* d_ws, size_t ws_size,
                              hipStream_t stream) {
  const float* x    = (const float*)d_in[0];   // [4,2048,1024]
  const float* Wqkv = (const float*)d_in[1];   // [3072,1024]
  const float* Wout = (const float*)d_in[2];   // [1024,1024]
  float* out = (float*)d_out;                  // [4,2048,1024] f32

  const size_t QK_ELEMS = (size_t)4 * 16 * 2048 * 64;   // 8388608
  __bf16* Qb = (__bf16*)d_ws;
  __bf16* Kb = Qb + QK_ELEMS;
  __bf16* Vt = Kb + QK_ELEMS;                  // transposed [b,h,d,s]
  __bf16* AO = Vt + QK_ELEMS;                  // attention out [8192,1024]

  // 1) QKV projection: [8192,1024] x [3072,1024]^T, scatter bf16 Q/K/Vt
  dim3 g1(8192 / 128, 3072 / 128);
  gemm_bf16_wmma<1, false><<<g1, 256, 0, stream>>>(
      x, nullptr, Wqkv, 1024, nullptr, 0, Qb, Kb, Vt);

  // 2) RoPE on Q and K
  int ropeThreads = 4 * 16 * 2048 * 32;
  rope_kernel<<<ropeThreads / 256, 256, 0, stream>>>(Qb, Kb);

  // 3) Causal flash attention
  attn_kernel<<<dim3(64, 2048 / 64), 128, 0, stream>>>(Qb, Kb, Vt, AO);

  // 4) Output projection: [8192,1024] x [1024,1024]^T -> f32 d_out
  dim3 g2(8192 / 128, 1024 / 128);
  gemm_bf16_wmma<0, true><<<g2, 256, 0, stream>>>(
      nullptr, AO, Wout, 1024, out, 1024, nullptr, nullptr, nullptr);
}
